// DeformableTransformerEncoderLayer_7267084665176
// MI455X (gfx1250) — matrix operand order repaired and checked by reading, not compiled, not run on hardware
//
#include <hip/hip_runtime.h>

typedef __attribute__((ext_vector_type(16))) __bf16 v16bf;
typedef __attribute__((ext_vector_type(8)))  __bf16 v8bf;
typedef __attribute__((ext_vector_type(8)))  float  v8f;

#define LQ      20197
#define NBATCH  2
#define RTOT    (NBATCH * LQ)          /* 40394 rows */
#define RTILES  ((RTOT + 15) / 16)     /* 2525 row tiles */
#define MBLKS   ((RTILES + 3) / 4)     /* 632 blocks of 4 row-tiles */

// -------------------------------------------------------------------------
// Pack a row-major f32 weight [K][N] into bf16 WMMA B-fragment order:
// per (ktile,ntile) a 512-element block; lane l holds column ntile*16+(l&15),
// K = ktile*32 + (l>>4)*8 + (i&7) + (i>>3)*16, i=0..15 (32B per lane).
// -------------------------------------------------------------------------
__global__ void pack_w(const float* __restrict__ W, __bf16* __restrict__ out,
                       int K, int N) {
  int t = blockIdx.x * 256 + threadIdx.x;
  if (t >= K * N) return;
  int ntiles = N >> 4;
  int b      = t >> 9;
  int within = t & 511;
  int lane   = within >> 4;
  int i      = within & 15;
  int kt     = b / ntiles;
  int nt     = b - kt * ntiles;
  int n      = (nt << 4) + (lane & 15);
  int k      = (kt << 5) + ((lane >> 4) << 3) + (i & 7) + ((i >> 3) << 4);
  out[t] = (__bf16)W[(size_t)k * N + n];
}

// -------------------------------------------------------------------------
// A-fragment loader: 16 bf16 of one row in WMMA A layout.
// lane<16: K = kb..kb+7 and kb+16..kb+23 of row (lane&15) (kb has +8 for hi half)
// ASRC: 0 = f32 (convert), 1 = f32 + f32 (fused add), 2 = bf16 direct.
// -------------------------------------------------------------------------
template<int ASRC>
__device__ __forceinline__ v16bf load_a_frag(const void* __restrict__ Aptr,
                                             const float* __restrict__ A2,
                                             size_t rowbase, int kb) {
  v16bf a;
  if (ASRC == 2) {
    const __bf16* A = (const __bf16*)Aptr;
    v8bf c0 = *(const v8bf*)(A + rowbase + kb);
    v8bf c1 = *(const v8bf*)(A + rowbase + kb + 16);
#pragma unroll
    for (int i = 0; i < 8; ++i) { a[i] = c0[i]; a[8 + i] = c1[i]; }
  } else {
    const float* A = (const float*)Aptr;
    v8f c0 = *(const v8f*)(A + rowbase + kb);
    v8f c1 = *(const v8f*)(A + rowbase + kb + 16);
    if (ASRC == 1) {
      c0 += *(const v8f*)(A2 + rowbase + kb);
      c1 += *(const v8f*)(A2 + rowbase + kb + 16);
    }
#pragma unroll
    for (int i = 0; i < 8; ++i) { a[i] = (__bf16)c0[i]; a[8 + i] = (__bf16)c1[i]; }
  }
  return a;
}

// -------------------------------------------------------------------------
// WMMA GEMM: C[R,N] = A[R,K] * Bsw + bias.
// One wave -> 16x64 tile (4 accumulators). K-loop unrolled x2 with
// disjoint A/B register sets so the 4 WMMAs of one phase overlap the
// A-conversion + B-loads of the other (no WAR hazard NOPs, batched loads).
// Block = 128 threads = 4 waves covering 64 rows; grid.y covers N/64.
// -------------------------------------------------------------------------
template<int ASRC, bool STORE_BF16, bool RELU>
__global__ void gemm_bf16_wmma(const void* __restrict__ Aptr,
                               const float* __restrict__ A2,
                               const __bf16* __restrict__ Bsw,
                               const float* __restrict__ bias,
                               void* __restrict__ Cptr,
                               int K, int N) {
  const int lane   = threadIdx.x & 31;
  const int wave   = threadIdx.x >> 5;
  const int mtile  = blockIdx.x * 4 + wave;
  const int ng0    = blockIdx.y * 4;                 // first of 4 n-tiles
  const int mrow   = (mtile << 4) + (lane & 15);
  const int rowc   = mrow < RTOT ? mrow : (RTOT - 1); // clamp, discard at store
  const int hsel   = lane >> 4;
  const int ntiles = N >> 4;
  const int ktiles = K >> 5;                          // always even (8 or 32)
  const size_t rowbase = (size_t)rowc * K;

  v8f acc[4] = {{}, {}, {}, {}};

  for (int kt = 0; kt < ktiles; kt += 2) {
    const int kb0 = (kt << 5) + (hsel << 3);
    const int kb1 = kb0 + 32;
    const __bf16* bp0 = Bsw + (((size_t)kt * ntiles + ng0) << 9) + ((size_t)lane << 4);
    const __bf16* bp1 = bp0 + ((size_t)ntiles << 9);

    v16bf a0 = load_a_frag<ASRC>(Aptr, A2, rowbase, kb0);
    v16bf a1 = load_a_frag<ASRC>(Aptr, A2, rowbase, kb1);

    v16bf b00 = *(const v16bf*)(bp0);
    v16bf b01 = *(const v16bf*)(bp0 + 512);
    v16bf b02 = *(const v16bf*)(bp0 + 1024);
    v16bf b03 = *(const v16bf*)(bp0 + 1536);
    v16bf b10 = *(const v16bf*)(bp1);
    v16bf b11 = *(const v16bf*)(bp1 + 512);
    v16bf b12 = *(const v16bf*)(bp1 + 1024);
    v16bf b13 = *(const v16bf*)(bp1 + 1536);

    acc[0] = __builtin_amdgcn_wmma_f32_16x16x32_bf16(false, a0, false, b00, (short)0, acc[0], false, false);
    acc[1] = __builtin_amdgcn_wmma_f32_16x16x32_bf16(false, a0, false, b01, (short)0, acc[1], false, false);
    acc[2] = __builtin_amdgcn_wmma_f32_16x16x32_bf16(false, a0, false, b02, (short)0, acc[2], false, false);
    acc[3] = __builtin_amdgcn_wmma_f32_16x16x32_bf16(false, a0, false, b03, (short)0, acc[3], false, false);
    acc[0] = __builtin_amdgcn_wmma_f32_16x16x32_bf16(false, a1, false, b10, (short)0, acc[0], false, false);
    acc[1] = __builtin_amdgcn_wmma_f32_16x16x32_bf16(false, a1, false, b11, (short)0, acc[1], false, false);
    acc[2] = __builtin_amdgcn_wmma_f32_16x16x32_bf16(false, a1, false, b12, (short)0, acc[2], false, false);
    acc[3] = __builtin_amdgcn_wmma_f32_16x16x32_bf16(false, a1, false, b13, (short)0, acc[3], false, false);
  }

#pragma unroll
  for (int j = 0; j < 4; ++j) {
    const int   col = ((ng0 + j) << 4) + (lane & 15);
    const float bv  = bias ? bias[col] : 0.0f;
#pragma unroll
    for (int rr = 0; rr < 8; ++rr) {
      int m = (mtile << 4) + rr + (hsel << 3);
      if (m < RTOT) {
        float v = acc[j][rr] + bv;
        if (RELU) v = v > 0.0f ? v : 0.0f;
        if (STORE_BF16) ((__bf16*)Cptr)[(size_t)m * N + col] = (__bf16)v;
        else            ((float*) Cptr)[(size_t)m * N + col] = v;
      }
    }
  }
}

// -------------------------------------------------------------------------
// Deformable attention sampling. Block = 256 = 8 waves; wave = head,
// lane = channel (D_HEAD = 32), so each corner gather is a coalesced
// 64-byte bf16 wave load. Softmax over 16 logits via wave32 shuffles.
// -------------------------------------------------------------------------
__global__ void msds_sample(const float* __restrict__ attnl,
                            const __bf16* __restrict__ offb,
                            const float* __restrict__ refp,
                            const __bf16* __restrict__ value,
                            __bf16* __restrict__ sampled) {
  const int r    = blockIdx.x;
  const int h    = threadIdx.x >> 5;
  const int lane = threadIdx.x & 31;
  const int n    = r / LQ;
  const int q    = r - n * LQ;

  float lv = (lane < 16) ? attnl[(size_t)r * 128 + h * 16 + lane] : -3.0e38f;
  float mx = lv;
#pragma unroll
  for (int o = 16; o; o >>= 1) { float t = __shfl_xor(mx, o, 32); mx = mx > t ? mx : t; }
  float e = (lane < 16) ? __expf(lv - mx) : 0.0f;
  float s = e;
#pragma unroll
  for (int o = 16; o; o >>= 1) s += __shfl_xor(s, o, 32);
  float wgt = e / s;

  const int Hs[4] = {100, 50, 25, 13};
  const int Ws[4] = {152, 76, 38, 19};
  const int St[4] = {0, 15200, 19000, 19950};

  float acc = 0.0f;
#pragma unroll
  for (int l = 0; l < 4; ++l) {
    const int   H  = Hs[l], W = Ws[l];
    const float rx = refp[(((size_t)(n * LQ + q)) * 4 + l) * 2 + 0];
    const float ry = refp[(((size_t)(n * LQ + q)) * 4 + l) * 2 + 1];
    const size_t vbase = (size_t)n * LQ + St[l];
#pragma unroll
    for (int p = 0; p < 4; ++p) {
      const size_t ob = (size_t)r * 256 + (size_t)(((h * 4 + l) * 4 + p) * 2);
      float ox = (float)offb[ob], oy = (float)offb[ob + 1];
      float aw = __shfl(wgt, l * 4 + p, 32);
      float x  = rx * (float)W + ox - 0.5f;
      float y  = ry * (float)H + oy - 0.5f;
      float xf = floorf(x), yf = floorf(y);
      float fx = x - xf,    fy = y - yf;
      int   x0 = (int)xf,   y0 = (int)yf;
#pragma unroll
      for (int dy = 0; dy < 2; ++dy)
#pragma unroll
        for (int dx = 0; dx < 2; ++dx) {
          int xi = x0 + dx, yi = y0 + dy;
          if (xi >= 0 && xi < W && yi >= 0 && yi < H) {
            float w = (dx ? fx : 1.0f - fx) * (dy ? fy : 1.0f - fy) * aw;
            acc += w * (float)value[(vbase + (size_t)yi * W + xi) * 256 + h * 32 + lane];
          }
        }
    }
  }
  sampled[(size_t)r * 256 + h * 32 + lane] = (__bf16)acc;
}

// -------------------------------------------------------------------------
// out = LayerNorm(a + b) * g + beta, one 256-thread block per row.
// -------------------------------------------------------------------------
__global__ void add_ln(const float* __restrict__ a, const float* __restrict__ b,
                       const float* __restrict__ g, const float* __restrict__ be,
                       float* __restrict__ out) {
  __shared__ float red[256];
  const int r = blockIdx.x;
  const int t = threadIdx.x;
  float x = a[(size_t)r * 256 + t] + b[(size_t)r * 256 + t];
  red[t] = x; __syncthreads();
  for (int o = 128; o; o >>= 1) { if (t < o) red[t] += red[t + o]; __syncthreads(); }
  float mean = red[0] * (1.0f / 256.0f); __syncthreads();
  float d = x - mean;
  red[t] = d * d; __syncthreads();
  for (int o = 128; o; o >>= 1) { if (t < o) red[t] += red[t + o]; __syncthreads(); }
  float var = red[0] * (1.0f / 256.0f);
  float rs  = rsqrtf(var + 1e-5f);
  out[(size_t)r * 256 + t] = d * rs * g[t] + be[t];
}

// -------------------------------------------------------------------------
extern "C" void kernel_launch(void* const* d_in, const int* in_sizes, int n_in,
                              void* d_out, int out_size, void* d_ws, size_t ws_size,
                              hipStream_t stream) {
  (void)in_sizes; (void)n_in; (void)out_size; (void)ws_size;
  const float* src     = (const float*)d_in[0];
  const float* pos     = (const float*)d_in[1];
  const float* refp    = (const float*)d_in[2];
  const float* w_value = (const float*)d_in[5];
  const float* b_value = (const float*)d_in[6];
  const float* w_off   = (const float*)d_in[7];
  const float* b_off   = (const float*)d_in[8];
  const float* w_attn  = (const float*)d_in[9];
  const float* b_attn  = (const float*)d_in[10];
  const float* w_out   = (const float*)d_in[11];
  const float* b_out   = (const float*)d_in[12];
  const float* g1      = (const float*)d_in[13];
  const float* beta1   = (const float*)d_in[14];
  const float* w1      = (const float*)d_in[15];
  const float* b1      = (const float*)d_in[16];
  const float* w2      = (const float*)d_in[17];
  const float* b2      = (const float*)d_in[18];
  const float* g2      = (const float*)d_in[19];
  const float* beta2   = (const float*)d_in[20];

  // ---- workspace layout (bytes) ----
  char* ws = (char*)d_ws;
  __bf16* wv_sw = (__bf16*)ws;            // 256x256
  __bf16* wo_sw = wv_sw + 65536;          // 256x256
  __bf16* wa_sw = wo_sw + 65536;          // 256x128
  __bf16* wp_sw = wa_sw + 32768;          // 256x256
  __bf16* w1_sw = wp_sw + 65536;          // 256x1024
  __bf16* w2_sw = w1_sw + 262144;         // 1024x256
  char* act = ws + 1507328;               // aligned activation block
  __bf16* value_b = (__bf16*)act;                       // [R,256] bf16  20.7MB
  __bf16* off_b   = (__bf16*)(act + 20681728);          // [R,256] bf16
  float*  attn_f  = (float*)(act + 41363456);           // [R,128] f32
  __bf16* sampled = (__bf16*)(act + 62045184);          // [R,256] bf16
  __bf16* h_b     = (__bf16*)act;                       // [R,1024] bf16 (aliases the 4 above, dead by FFN1)
  float*  src2y   = (float*)(act + 82726912);           // [R,256] f32 (src2, later FFN2 out)
  float*  x1      = (float*)(act + 82726912 + 41363456);// [R,256] f32

  // ---- pack weights to bf16 fragment layout (tiny, once per call) ----
  pack_w<<<256,  256, 0, stream>>>(w_value, wv_sw, 256, 256);
  pack_w<<<256,  256, 0, stream>>>(w_off,   wo_sw, 256, 256);
  pack_w<<<128,  256, 0, stream>>>(w_attn,  wa_sw, 256, 128);
  pack_w<<<256,  256, 0, stream>>>(w_out,   wp_sw, 256, 256);
  pack_w<<<1024, 256, 0, stream>>>(w1,      w1_sw, 256, 1024);
  pack_w<<<1024, 256, 0, stream>>>(w2,      w2_sw, 1024, 256);

  dim3 blk(128);
  // value = src @ Wv + b  (bf16 out)
  gemm_bf16_wmma<0, true,  false><<<dim3(MBLKS, 4),  blk, 0, stream>>>(src, nullptr, wv_sw, b_value, value_b, 256, 256);
  // off = (src+pos) @ Woff + b  (bf16 out)
  gemm_bf16_wmma<1, true,  false><<<dim3(MBLKS, 4),  blk, 0, stream>>>(src, pos,     wo_sw, b_off,   off_b,   256, 256);
  // attn logits = (src+pos) @ Wattn + b  (f32 out)
  gemm_bf16_wmma<1, false, false><<<dim3(MBLKS, 2),  blk, 0, stream>>>(src, pos,     wa_sw, b_attn,  attn_f,  256, 128);
  // softmax + bilinear sample + point-weighted sum
  msds_sample<<<RTOT, 256, 0, stream>>>(attn_f, off_b, refp, value_b, sampled);
  // src2 = sampled @ Wout + b  (f32 out)
  gemm_bf16_wmma<2, false, false><<<dim3(MBLKS, 4),  blk, 0, stream>>>(sampled, nullptr, wp_sw, b_out, src2y, 256, 256);
  // x1 = LN(src + src2)
  add_ln<<<RTOT, 256, 0, stream>>>(src, src2y, g1, beta1, x1);
  // h = relu(x1 @ W1 + b1)  (bf16 out, aliases early buffers)
  gemm_bf16_wmma<0, true,  true ><<<dim3(MBLKS, 16), blk, 0, stream>>>(x1, nullptr, w1_sw, b1, h_b, 256, 1024);
  // y = h @ W2 + b2  (f32 out, reuses src2 buffer)
  gemm_bf16_wmma<2, false, false><<<dim3(MBLKS, 4),  blk, 0, stream>>>(h_b, nullptr, w2_sw, b2, src2y, 1024, 256);
  // out = LN(x1 + y)
  add_ln<<<RTOT, 256, 0, stream>>>(x1, src2y, g2, beta2, (float*)d_out);
}
